// MegNetBlock_52209622450459
// MI455X (gfx1250) — matrix operand order repaired and checked by reading, not compiled, not run on hardware
//
#include <hip/hip_runtime.h>
#include <hip/hip_bf16.h>

typedef __attribute__((ext_vector_type(16))) _Float16 v16h;
typedef __attribute__((ext_vector_type(8)))  float    v8f;
typedef __attribute__((ext_vector_type(4)))  float    f32x4;

// Branch-free fast softplus built on the raw v_exp_f32 / v_log_f32 ops.
// softplus(x) = max(x,0) + ln2 * log2(1 + exp2(-|x|*log2e)).
// Log argument is in (1,2] so no denorm guard is needed -> ~6 VALU ops.
__device__ __forceinline__ float softplus_f(float x) {
    const float LOG2E = 1.4426950408889634f;
    const float LN2   = 0.6931471805599453f;
    float t = __builtin_amdgcn_exp2f(-fabsf(x) * LOG2E);
    return fmaxf(x, 0.f) + LN2 * __builtin_amdgcn_logf(1.0f + t);
}

// One 16x16 output tile of  X[64xK] (f16, LDS, row-major, leading dim ldx)
//   times Wt (f16, global, stored transposed [Ncols][K])  + bias, f32 accum.
// Layouts per CDNA5 ISA 7.12.2 (wave32):
//   A (16xK f16): lane row = lane&15 ; elems 0..7 -> K = k0 + (lane>>4)*8 + e
//                 elems 8..15 -> K = k0 + 16 + (lane>>4)*8 + e
//   B (Kx16 f16): lane col = lane&15 ; same K mapping as A (table order)
//   C/D f32:      elem r -> row m0 + r + 8*(lane>>4), col n0 + (lane&15)
__device__ __forceinline__ v8f tile_mma(const _Float16* __restrict__ X, int ldx,
                                        const _Float16* __restrict__ Wt, int K,
                                        const float* __restrict__ bias,
                                        int m0, int n0, int lane) {
    const int col = n0 + (lane & 15);
    const int row = m0 + (lane & 15);
    const int hi  = lane >> 4;
    const float bv = bias[col];
    v8f c = {bv, bv, bv, bv, bv, bv, bv, bv};
#pragma unroll
    for (int k0 = 0; k0 < K; k0 += 32) {
        v16h a, b;
        const _Float16* pa = X + row * ldx + k0 + hi * 8;
#pragma unroll
        for (int e = 0; e < 8; ++e)  a[e]     = pa[e];
#pragma unroll
        for (int e = 0; e < 8; ++e)  a[8 + e] = pa[16 + e];
        const _Float16* pb = Wt + (size_t)col * K + k0 + hi * 8;
#pragma unroll
        for (int e = 0; e < 8; ++e)  b[e]     = pb[e];
#pragma unroll
        for (int e = 0; e < 8; ++e)  b[8 + e] = pb[16 + e];
        c = __builtin_amdgcn_wmma_f32_16x16x32_f16(false, a, false, b,
                                                   (short)0, c, false, false);
    }
    return c;
}

// softplus + f16 store of a C tile into LDS buffer Y (leading dim ldy)
__device__ __forceinline__ void store_sp_f16(v8f c, _Float16* __restrict__ Y, int ldy,
                                             int m0, int n0, int lane) {
    const int col   = n0 + (lane & 15);
    const int rbase = m0 + ((lane >> 4) * 8);
#pragma unroll
    for (int r = 0; r < 8; ++r)
        Y[(rbase + r) * ldy + col] = (_Float16)softplus_f(c[r]);
}

// softplus + f32 store of a C tile into LDS buffer Y (leading dim 64)
__device__ __forceinline__ void store_sp_f32(v8f c, float* __restrict__ Y,
                                             int m0, int n0, int lane) {
    const int col   = n0 + (lane & 15);
    const int rbase = m0 + ((lane >> 4) * 8);
#pragma unroll
    for (int r = 0; r < 8; ++r)
        Y[(rbase + r) * 64 + col] = softplus_f(c[r]);
}

// ---------- weight convert + transpose: W[K][N] f32 -> Wt[N][K] f16 ----------
__global__ void k_wt(const float* __restrict__ W, _Float16* __restrict__ Wt,
                     int K, int N) {
    int i = blockIdx.x * blockDim.x + threadIdx.x;
    if (i < K * N) {
        int k = i / N, n = i % N;
        Wt[(size_t)n * K + k] = (_Float16)W[i];
    }
}

// ---------- pre-dense attr: u_pre = softplus(u @ W + b), 1x64 ----------
__global__ void k_pre_attr(const float* __restrict__ u_in, const float* __restrict__ W,
                           const float* __restrict__ b, float* __restrict__ u_pre) {
    int j = threadIdx.x; // 64 threads
    float acc = b[j];
    for (int k = 0; k < 64; ++k) acc += u_in[k] * W[k * 64 + j];
    u_pre[j] = softplus_f(acc);
}

// ---------- pre-dense nodes: v_pre = softplus(node_feat @ W + b), f16 out ----
__global__ void __launch_bounds__(256)
k_pre_node(const float* __restrict__ node_feat, const _Float16* __restrict__ Wt,
           const float* __restrict__ bias, _Float16* __restrict__ v_pre, int N) {
    __shared__ _Float16 sIn[64 * 64];
    __shared__ _Float16 sOut[64 * 64];
    const int tid = threadIdx.x, wave = tid >> 5, lane = tid & 31;
    const int base = blockIdx.x * 64;
    for (int idx = tid; idx < 64 * 64; idx += 256) {
        int r = idx >> 6, c2 = idx & 63;
        int n = base + r; if (n >= N) n = N - 1;
        sIn[idx] = (_Float16)node_feat[(size_t)n * 64 + c2];
    }
    __syncthreads();
#pragma unroll
    for (int t0 = 0; t0 < 2; ++t0) {
        int t = wave + t0 * 8;
        int m0 = (t >> 2) * 16, n0 = (t & 3) * 16;
        v8f c = tile_mma(sIn, 64, Wt, 64, bias, m0, n0, lane);
        store_sp_f16(c, sOut, 64, m0, n0, lane);
    }
    __syncthreads();
    // coalesced 16B stores, uniform bound (no per-lane branching)
    const int rmax = min(64, N - base);
    for (int idx = tid; idx < rmax * 8; idx += 256) {
        int r = idx >> 3, ch = idx & 7;
        ((uint4*)(v_pre + (size_t)(base + r) * 64))[ch] = ((const uint4*)(sOut + r * 64))[ch];
    }
}

// ---------- fused edge pipeline ----------
// per block: 64 edges, 8 waves. X = [v_src | v_dst | e_pre | u] (64x256 f16)
__global__ void __launch_bounds__(256)
k_edge(const float* __restrict__ edge_feat, const int* __restrict__ src,
       const int* __restrict__ dst, const float* __restrict__ u_pre,
       const _Float16* __restrict__ v_pre,
       const _Float16* __restrict__ WtPreE, const float* __restrict__ bPreE,
       const _Float16* __restrict__ Wt0, const float* __restrict__ b0,
       const _Float16* __restrict__ Wt1, const float* __restrict__ b1,
       const _Float16* __restrict__ Wt2, const float* __restrict__ b2,
       float* __restrict__ e_out, float* __restrict__ ve_sum,
       int* __restrict__ cnt, float* __restrict__ esum, int E) {
    __shared__ _Float16 sX[64 * 256];   // 32 KB
    __shared__ _Float16 sY[64 * 128];   // 16 KB (also reused as float[64*64])
    __shared__ int sSrc[64];
    __shared__ int sDst[64];
    const int tid = threadIdx.x, wave = tid >> 5, lane = tid & 31;
    const int base = blockIdx.x * 64;

    if (tid < 64) {
        int gi = base + tid; int gic = gi < E ? gi : E - 1;
        sSrc[tid] = src[gic];
        sDst[tid] = dst[gic];
    }
    __syncthreads();
    // gather v_pre[src], v_pre[dst] as 16B chunks (8 halves) into cols 0..127
    for (int idx = tid; idx < 64 * 8; idx += 256) {
        int r = idx >> 3, ch = idx & 7;
        int s = sSrc[r], d = sDst[r];
        ((uint4*)(sX + r * 256))[ch]      = ((const uint4*)(v_pre + (size_t)s * 64))[ch];
        ((uint4*)(sX + r * 256 + 64))[ch] = ((const uint4*)(v_pre + (size_t)d * 64))[ch];
    }
    // raw edge feats -> sY (64x64 staging), u_pre broadcast -> cols 192..255
    for (int idx = tid; idx < 64 * 64; idx += 256) {
        int r = idx >> 6, c2 = idx & 63;
        int gi = base + r; int gic = gi < E ? gi : E - 1;
        sY[r * 64 + c2]        = (_Float16)edge_feat[(size_t)gic * 64 + c2];
        sX[r * 256 + 192 + c2] = (_Float16)u_pre[c2];
    }
    __syncthreads();

    // fused pre-dense edge layer: e_pre = softplus(raw @ WpreE + b) -> cols 128..191
#pragma unroll
    for (int t0 = 0; t0 < 2; ++t0) {
        int t = wave + t0 * 8;
        int m0 = (t >> 2) * 16, n0 = (t & 3) * 16;
        v8f c = tile_mma(sY, 64, WtPreE, 64, bPreE, m0, n0, lane);
        store_sp_f16(c, sX + 128, 256, m0, n0, lane);
    }
    __syncthreads();

    // layer0: [64x256] @ [256x128] -> sY
#pragma unroll
    for (int t0 = 0; t0 < 4; ++t0) {
        int t = wave + t0 * 8;
        int m0 = (t >> 3) * 16, n0 = (t & 7) * 16;
        v8f c = tile_mma(sX, 256, Wt0, 256, b0, m0, n0, lane);
        store_sp_f16(c, sY, 128, m0, n0, lane);
    }
    __syncthreads();

    // layer1: [64x128] @ [128x128] -> sZ (reuse sX as 64x128)
    _Float16* sZ = sX;
#pragma unroll
    for (int t0 = 0; t0 < 4; ++t0) {
        int t = wave + t0 * 8;
        int m0 = (t >> 3) * 16, n0 = (t & 7) * 16;
        v8f c = tile_mma(sY, 128, Wt1, 128, b1, m0, n0, lane);
        store_sp_f16(c, sZ, 128, m0, n0, lane);
    }
    __syncthreads();

    // layer2: [64x128] @ [128x64] -> e_new staged f32 in LDS (sY reused)
    float* sEnew = (float*)sY;   // 64*64 f32 == 16 KB
#pragma unroll
    for (int t0 = 0; t0 < 2; ++t0) {
        int t = wave + t0 * 8;
        int m0 = (t >> 2) * 16, n0 = (t & 3) * 16;
        v8f c = tile_mma(sZ, 128, Wt2, 128, b2, m0, n0, lane);
        store_sp_f32(c, sEnew, m0, n0, lane);
    }
    __syncthreads();

    // coalesced epilogue: b128 skip-add + NT store, segment atomics
    const int rmax = min(64, E - base);
    for (int idx = tid; idx < rmax * 16; idx += 256) {
        int r = idx >> 4, ch = idx & 15;
        size_t goff = (size_t)(base + r) * 64 + ch * 4;
        f32x4 en = *(const f32x4*)&sEnew[r * 64 + ch * 4];
        f32x4 ef = *(const f32x4*)&edge_feat[goff];
        __builtin_nontemporal_store(en + ef, (f32x4*)&e_out[goff]);
        float* vp = &ve_sum[(size_t)sDst[r] * 64 + ch * 4];
        atomicAdd(vp + 0, en.x);
        atomicAdd(vp + 1, en.y);
        atomicAdd(vp + 2, en.z);
        atomicAdd(vp + 3, en.w);
    }
    // per-column reduction for the edge-mean readout + dst histogram
    if (tid < 64) {
        float s = 0.f;
        for (int r = 0; r < rmax; ++r) s += sEnew[r * 64 + tid];
        atomicAdd(&esum[tid], s);
        if (tid < rmax) atomicAdd(&cnt[sDst[tid]], 1);
    }
}

// ---------- fused node pipeline: X = [v_pre | ve_mean | u] (64x192) ----------
__global__ void __launch_bounds__(256)
k_node(const float* __restrict__ node_feat, const _Float16* __restrict__ v_pre,
       const float* __restrict__ ve_sum, const int* __restrict__ cnt,
       const float* __restrict__ u_pre,
       const _Float16* __restrict__ Wt0, const float* __restrict__ b0,
       const _Float16* __restrict__ Wt1, const float* __restrict__ b1,
       const _Float16* __restrict__ Wt2, const float* __restrict__ b2,
       float* __restrict__ v_out, float* __restrict__ vsum, int N) {
    __shared__ _Float16 sX[64 * 192];   // 24 KB
    __shared__ _Float16 sY[64 * 128];   // 16 KB (reused as float[64*64])
    const int tid = threadIdx.x, wave = tid >> 5, lane = tid & 31;
    const int base = blockIdx.x * 64;
    for (int idx = tid; idx < 64 * 64; idx += 256) {
        int r = idx >> 6, c2 = idx & 63;
        int n = base + r; int ncl = n < N ? n : N - 1;
        sX[r * 192 + c2] = v_pre[(size_t)ncl * 64 + c2];
        float cv = (float)cnt[ncl]; if (cv < 1.f) cv = 1.f;
        sX[r * 192 + 64 + c2]  = (_Float16)(ve_sum[(size_t)ncl * 64 + c2] / cv);
        sX[r * 192 + 128 + c2] = (_Float16)u_pre[c2];
    }
    __syncthreads();
#pragma unroll
    for (int t0 = 0; t0 < 4; ++t0) {
        int t = wave + t0 * 8;
        int m0 = (t >> 3) * 16, n0 = (t & 7) * 16;
        v8f c = tile_mma(sX, 192, Wt0, 192, b0, m0, n0, lane);
        store_sp_f16(c, sY, 128, m0, n0, lane);
    }
    __syncthreads();
    _Float16* sZ = sX;
#pragma unroll
    for (int t0 = 0; t0 < 4; ++t0) {
        int t = wave + t0 * 8;
        int m0 = (t >> 3) * 16, n0 = (t & 7) * 16;
        v8f c = tile_mma(sY, 128, Wt1, 128, b1, m0, n0, lane);
        store_sp_f16(c, sZ, 128, m0, n0, lane);
    }
    __syncthreads();
    float* sVnew = (float*)sY;   // 64*64 f32
#pragma unroll
    for (int t0 = 0; t0 < 2; ++t0) {
        int t = wave + t0 * 8;
        int m0 = (t >> 2) * 16, n0 = (t & 3) * 16;
        v8f c = tile_mma(sZ, 128, Wt2, 128, b2, m0, n0, lane);
        store_sp_f32(c, sVnew, m0, n0, lane);
    }
    __syncthreads();
    const int rmax = min(64, N - base);
    for (int idx = tid; idx < rmax * 16; idx += 256) {
        int r = idx >> 4, ch = idx & 15;
        size_t goff = (size_t)(base + r) * 64 + ch * 4;
        f32x4 vn = *(const f32x4*)&sVnew[r * 64 + ch * 4];
        f32x4 vf = *(const f32x4*)&node_feat[goff];
        __builtin_nontemporal_store(vn + vf, (f32x4*)&v_out[goff]);
    }
    if (tid < 64) {
        float s = 0.f;
        for (int r = 0; r < rmax; ++r) s += sVnew[r * 64 + tid];
        atomicAdd(&vsum[tid], s);
    }
}

// ---------- attr update (1x192 MLP, scalar) ----------
__global__ void k_attr(const float* __restrict__ u_in, const float* __restrict__ u_pre,
                       const float* __restrict__ esum, const float* __restrict__ vsum,
                       const float* __restrict__ W0, const float* __restrict__ b0,
                       const float* __restrict__ W1, const float* __restrict__ b1,
                       const float* __restrict__ W2, const float* __restrict__ b2,
                       float* __restrict__ u_out, int E, int N) {
    __shared__ float x[192], h1[128], h2[128];
    int tid = threadIdx.x; // 128 threads
    if (tid < 64) {
        x[tid]       = u_pre[tid];
        x[64 + tid]  = esum[tid] / (float)E;
        x[128 + tid] = vsum[tid] / (float)N;
    }
    __syncthreads();
    float acc = b0[tid];
    for (int k = 0; k < 192; ++k) acc += x[k] * W0[k * 128 + tid];
    h1[tid] = softplus_f(acc);
    __syncthreads();
    acc = b1[tid];
    for (int k = 0; k < 128; ++k) acc += h1[k] * W1[k * 128 + tid];
    h2[tid] = softplus_f(acc);
    __syncthreads();
    if (tid < 64) {
        float a2 = b2[tid];
        for (int k = 0; k < 128; ++k) a2 += h2[k] * W2[k * 64 + tid];
        u_out[tid] = softplus_f(a2) + u_in[tid];
    }
}

extern "C" void kernel_launch(void* const* d_in, const int* in_sizes, int n_in,
                              void* d_out, int out_size, void* d_ws, size_t ws_size,
                              hipStream_t stream) {
    const float* edge_feat  = (const float*)d_in[0];
    const float* node_feat  = (const float*)d_in[1];
    const float* graph_attr = (const float*)d_in[2];
    const int*   src        = (const int*)d_in[3];
    const int*   dst        = (const int*)d_in[4];
    const float* pre_edge_w = (const float*)d_in[5];
    const float* pre_edge_b = (const float*)d_in[6];
    const float* pre_node_w = (const float*)d_in[7];
    const float* pre_node_b = (const float*)d_in[8];
    const float* pre_attr_w = (const float*)d_in[9];
    const float* pre_attr_b = (const float*)d_in[10];
    const float* edge_w0 = (const float*)d_in[11]; const float* edge_b0 = (const float*)d_in[12];
    const float* edge_w1 = (const float*)d_in[13]; const float* edge_b1 = (const float*)d_in[14];
    const float* edge_w2 = (const float*)d_in[15]; const float* edge_b2 = (const float*)d_in[16];
    const float* node_w0 = (const float*)d_in[17]; const float* node_b0 = (const float*)d_in[18];
    const float* node_w1 = (const float*)d_in[19]; const float* node_b1 = (const float*)d_in[20];
    const float* node_w2 = (const float*)d_in[21]; const float* node_b2 = (const float*)d_in[22];
    const float* attr_w0 = (const float*)d_in[23]; const float* attr_b0 = (const float*)d_in[24];
    const float* attr_w1 = (const float*)d_in[25]; const float* attr_b1 = (const float*)d_in[26];
    const float* attr_w2 = (const float*)d_in[27]; const float* attr_b2 = (const float*)d_in[28];

    const int E = in_sizes[3];
    const int N = in_sizes[1] / 64;

    float* e_out = (float*)d_out;
    float* v_out = e_out + (size_t)E * 64;
    float* u_out = v_out + (size_t)N * 64;

    // workspace layout
    char* ws = (char*)d_ws;
    size_t off = 0;
    auto take = [&](size_t bytes) -> void* {
        void* p = ws + off;
        off = (off + bytes + 255) & ~(size_t)255;
        return p;
    };
    _Float16* WtPreE = (_Float16*)take(64 * 64 * 2);
    _Float16* WtPreN = (_Float16*)take(64 * 64 * 2);
    _Float16* WtE0   = (_Float16*)take(128 * 256 * 2);
    _Float16* WtE1   = (_Float16*)take(128 * 128 * 2);
    _Float16* WtE2   = (_Float16*)take(64 * 128 * 2);
    _Float16* WtN0   = (_Float16*)take(128 * 192 * 2);
    _Float16* WtN1   = (_Float16*)take(128 * 128 * 2);
    _Float16* WtN2   = (_Float16*)take(64 * 128 * 2);
    float*    u_pre  = (float*)take(64 * 4);
    _Float16* v_pre  = (_Float16*)take((size_t)N * 64 * 2);
    size_t zstart = off;
    float* ve_sum = (float*)take((size_t)N * 64 * 4);
    int*   cnt    = (int*)take((size_t)N * 4);
    float* esum   = (float*)take(64 * 4);
    float* vsum   = (float*)take(64 * 4);
    size_t zend = off;

    hipMemsetAsync(ws + zstart, 0, zend - zstart, stream);

    auto tconv = [&](const float* W, _Float16* Wt, int K, int Nc) {
        int total = K * Nc;
        k_wt<<<(total + 255) / 256, 256, 0, stream>>>(W, Wt, K, Nc);
    };
    tconv(pre_edge_w, WtPreE, 64, 64);
    tconv(pre_node_w, WtPreN, 64, 64);
    tconv(edge_w0, WtE0, 256, 128);
    tconv(edge_w1, WtE1, 128, 128);
    tconv(edge_w2, WtE2, 128, 64);
    tconv(node_w0, WtN0, 192, 128);
    tconv(node_w1, WtN1, 128, 128);
    tconv(node_w2, WtN2, 128, 64);

    k_pre_attr<<<1, 64, 0, stream>>>(graph_attr, pre_attr_w, pre_attr_b, u_pre);
    k_pre_node<<<(N + 63) / 64, 256, 0, stream>>>(node_feat, WtPreN, pre_node_b, v_pre, N);

    k_edge<<<(E + 63) / 64, 256, 0, stream>>>(edge_feat, src, dst, u_pre, v_pre,
                                              WtPreE, pre_edge_b,
                                              WtE0, edge_b0, WtE1, edge_b1, WtE2, edge_b2,
                                              e_out, ve_sum, cnt, esum, E);

    k_node<<<(N + 63) / 64, 256, 0, stream>>>(node_feat, v_pre, ve_sum, cnt, u_pre,
                                              WtN0, node_b0, WtN1, node_b1, WtN2, node_b2,
                                              v_out, vsum, N);

    k_attr<<<1, 128, 0, stream>>>(graph_attr, u_pre, esum, vsum,
                                  attr_w0, attr_b0, attr_w1, attr_b1, attr_w2, attr_b2,
                                  u_out, E, N);
}
